// SpatialGATAutoencoder_70342974373914
// MI455X (gfx1250) — matrix operand order repaired and checked by reading, not compiled, not run on hardware
//
#include <hip/hip_runtime.h>

#define N_NODES 20000
#define N_EDGES 320000
#define IN_CH   512
#define HIDT    512   // HEADS*HID
#define HEADS   4
#define HID     128
#define OUT_CH  30
#define OUT_P   32    // padded
#define NEG_SLOPE 0.2f

typedef __bf16 bf16_t;
typedef __attribute__((ext_vector_type(16))) __bf16 v16bf;
typedef __attribute__((ext_vector_type(8)))  __bf16 v8bf;
typedef __attribute__((ext_vector_type(8)))  float  v8f;

// ---------------- helpers ----------------
__device__ __forceinline__ v8bf ld8(const bf16_t* p) {
    return *reinterpret_cast<const v8bf*>(p);
}
__device__ __forceinline__ v16bf mk16(v8bf lo, v8bf hi) {
    v16bf r;
#pragma unroll
    for (int i = 0; i < 8; ++i) { r[i] = lo[i]; r[i + 8] = hi[i]; }
    return r;
}
__device__ __forceinline__ float lrelu(float x) { return x > 0.f ? x : NEG_SLOPE * x; }
// order-preserving float->uint encoding for atomicMax
__device__ __forceinline__ unsigned fenc(float f) {
    unsigned u = __float_as_uint(f);
    return (u & 0x80000000u) ? ~u : (u | 0x80000000u);
}
__device__ __forceinline__ float fdec(unsigned u) {
    unsigned v = (u & 0x80000000u) ? (u & 0x7fffffffu) : ~u;
    return __uint_as_float(v);
}

// ---------------- pack / convert ----------------
__global__ void k_cvt_x(const float* __restrict__ x, bf16_t* __restrict__ xbf) {
    int idx = blockIdx.x * blockDim.x + threadIdx.x;           // N*512 exact
    xbf[idx] = (bf16_t)x[idx];
}
__global__ void k_pack_w1t(const float* __restrict__ W1, bf16_t* __restrict__ w1t) {
    int idx = blockIdx.x * blockDim.x + threadIdx.x;           // 512*512 exact
    int n = idx >> 9, k = idx & 511;
    w1t[(size_t)n * IN_CH + k] = (bf16_t)W1[(size_t)k * HIDT + n];
}
__global__ void k_pack_w2t(const float* __restrict__ W2, bf16_t* __restrict__ w2t) {
    int idx = blockIdx.x * blockDim.x + threadIdx.x;           // 32*512 exact
    int n = idx >> 9, k = idx & 511;
    float v = (n < OUT_CH) ? W2[(size_t)k * OUT_CH + n] : 0.f;
    w2t[(size_t)n * HIDT + k] = (bf16_t)v;
}
__global__ void k_pack_wdt(const float* __restrict__ Wd, bf16_t* __restrict__ wdt) {
    int idx = blockIdx.x * blockDim.x + threadIdx.x;           // 512*32 exact
    int j = idx >> 5, k = idx & 31;
    float v = (k < OUT_CH) ? Wd[(size_t)k * IN_CH + j] : 0.f;
    wdt[(size_t)j * OUT_P + k] = (bf16_t)v;
}

// ---------------- WMMA GEMMs ----------------
// h1[N,512] = x[N,512] @ W1[512,512]   (A: xbf row-major, B: w1t col-packed)
__global__ __launch_bounds__(128) void k_gemm1(const bf16_t* __restrict__ A,
                                               const bf16_t* __restrict__ Bp,
                                               float* __restrict__ C) {
    const int lane = threadIdx.x & 31;
    const int wave = threadIdx.x >> 5;
    const int m0 = blockIdx.x << 4;
    const int n0 = ((blockIdx.y << 2) + wave) << 4;
    const int mm = lane & 15, g = lane >> 4;
    const bf16_t* arow = A + (size_t)(m0 + mm) * IN_CH;
    const bf16_t* bcol = Bp + (size_t)(n0 + mm) * IN_CH;
    v8f acc = {};
    for (int k0 = 0; k0 < IN_CH; k0 += 32) {
        v16bf a = mk16(ld8(arow + k0 + 8 * g), ld8(arow + k0 + 16 + 8 * g));
        v16bf b = mk16(ld8(bcol + k0 + 8 * g), ld8(bcol + k0 + 16 + 8 * g));
        acc = __builtin_amdgcn_wmma_f32_16x16x32_bf16(false, a, false, b,
                                                      (short)0, acc, false, false);
    }
#pragma unroll
    for (int r = 0; r < 8; ++r)
        C[(size_t)(m0 + r + 8 * g) * HIDT + n0 + mm] = acc[r];
}

// zpre[N,30] = hact[N,512] @ W2[512,30]  (padded to 32 cols)
__global__ __launch_bounds__(64) void k_gemm2(const bf16_t* __restrict__ A,
                                              const bf16_t* __restrict__ Bp,
                                              float* __restrict__ Z) {
    const int lane = threadIdx.x & 31;
    const int wave = threadIdx.x >> 5;
    const int m0 = blockIdx.x << 4;
    const int n0 = wave << 4;
    const int mm = lane & 15, g = lane >> 4;
    const bf16_t* arow = A + (size_t)(m0 + mm) * HIDT;
    const bf16_t* bcol = Bp + (size_t)(n0 + mm) * HIDT;
    v8f acc = {};
    for (int k0 = 0; k0 < HIDT; k0 += 32) {
        v16bf a = mk16(ld8(arow + k0 + 8 * g), ld8(arow + k0 + 16 + 8 * g));
        v16bf b = mk16(ld8(bcol + k0 + 8 * g), ld8(bcol + k0 + 16 + 8 * g));
        acc = __builtin_amdgcn_wmma_f32_16x16x32_bf16(false, a, false, b,
                                                      (short)0, acc, false, false);
    }
    const int col = n0 + mm;
    if (col < OUT_CH) {
#pragma unroll
        for (int r = 0; r < 8; ++r)
            Z[(size_t)(m0 + r + 8 * g) * OUT_CH + col] = acc[r];
    }
}

// x_hat[N,512] = z[N,32pad] @ Wd[32pad,512] + bd  (single K step)
__global__ __launch_bounds__(128) void k_gemm3(const bf16_t* __restrict__ A,
                                               const bf16_t* __restrict__ Bp,
                                               const float* __restrict__ bd,
                                               float* __restrict__ Xh) {
    const int lane = threadIdx.x & 31;
    const int wave = threadIdx.x >> 5;
    const int m0 = blockIdx.x << 4;
    const int n0 = ((blockIdx.y << 2) + wave) << 4;
    const int mm = lane & 15, g = lane >> 4;
    const bf16_t* arow = A + (size_t)(m0 + mm) * OUT_P;
    const bf16_t* bcol = Bp + (size_t)(n0 + mm) * OUT_P;
    v16bf a = mk16(ld8(arow + 8 * g), ld8(arow + 16 + 8 * g));
    v16bf b = mk16(ld8(bcol + 8 * g), ld8(bcol + 16 + 8 * g));
    v8f acc = {};
    acc = __builtin_amdgcn_wmma_f32_16x16x32_bf16(false, a, false, b,
                                                  (short)0, acc, false, false);
    const float bias = bd[n0 + mm];
#pragma unroll
    for (int r = 0; r < 8; ++r)
        Xh[(size_t)(m0 + r + 8 * g) * IN_CH + n0 + mm] = acc[r] + bias;
}

// ---------------- layer-1 attention ----------------
__global__ void k_attn1(const float* __restrict__ h1, const float* __restrict__ a_s,
                        const float* __restrict__ a_d,
                        float* __restrict__ als, float* __restrict__ ald) {
    int node = blockIdx.x * (blockDim.x >> 5) + (threadIdx.x >> 5);
    int lane = threadIdx.x & 31;
    if (node >= N_NODES) return;
    const float* hp = h1 + (size_t)node * HIDT;
    float ps[HEADS], pd[HEADS];
#pragma unroll
    for (int h = 0; h < HEADS; ++h) { ps[h] = 0.f; pd[h] = 0.f; }
#pragma unroll
    for (int h = 0; h < HEADS; ++h)
#pragma unroll
        for (int j = 0; j < 4; ++j) {
            int ch = h * HID + j * 32 + lane;
            float v = hp[ch];
            ps[h] += v * a_s[ch];
            pd[h] += v * a_d[ch];
        }
    for (int off = 16; off; off >>= 1)
#pragma unroll
        for (int h = 0; h < HEADS; ++h) {
            ps[h] += __shfl_xor(ps[h], off, 32);
            pd[h] += __shfl_xor(pd[h], off, 32);
        }
    if (lane == 0)
#pragma unroll
        for (int h = 0; h < HEADS; ++h) {
            als[node * HEADS + h] = ps[h];
            ald[node * HEADS + h] = pd[h];
        }
}

__global__ void k_init_max1(const float* __restrict__ als, const float* __restrict__ ald,
                            unsigned* __restrict__ umax) {
    int idx = blockIdx.x * blockDim.x + threadIdx.x;
    if (idx >= N_NODES * HEADS) return;
    umax[idx] = fenc(lrelu(als[idx] + ald[idx]));      // self-loop init
}
__global__ void k_edge_max1(const int* __restrict__ ei, const float* __restrict__ als,
                            const float* __restrict__ ald, unsigned* __restrict__ umax) {
    int e = blockIdx.x * blockDim.x + threadIdx.x;
    if (e >= N_EDGES) return;
    int s = ei[e], d = ei[N_EDGES + e];
#pragma unroll
    for (int h = 0; h < HEADS; ++h) {
        float v = lrelu(als[s * HEADS + h] + ald[d * HEADS + h]);
        atomicMax(&umax[d * HEADS + h], fenc(v));
    }
}
__global__ void k_sum_init1(const unsigned* __restrict__ umax, const float* __restrict__ als,
                            const float* __restrict__ ald, float* __restrict__ maxf,
                            float* __restrict__ sum) {
    int idx = blockIdx.x * blockDim.x + threadIdx.x;
    if (idx >= N_NODES * HEADS) return;
    float m = fdec(umax[idx]);
    maxf[idx] = m;
    sum[idx] = __expf(lrelu(als[idx] + ald[idx]) - m);  // self-loop term
}
__global__ void k_edge_sum1(const int* __restrict__ ei, const float* __restrict__ als,
                            const float* __restrict__ ald, const float* __restrict__ maxf,
                            float* __restrict__ sum) {
    int e = blockIdx.x * blockDim.x + threadIdx.x;
    if (e >= N_EDGES) return;
    int s = ei[e], d = ei[N_EDGES + e];
#pragma unroll
    for (int h = 0; h < HEADS; ++h) {
        float v = lrelu(als[s * HEADS + h] + ald[d * HEADS + h]);
        atomicAdd(&sum[d * HEADS + h], __expf(v - maxf[d * HEADS + h]));
    }
}
__global__ void k_agg_init1(const float* __restrict__ h1, const float* __restrict__ als,
                            const float* __restrict__ ald, const float* __restrict__ maxf,
                            const float* __restrict__ sum, float* __restrict__ out1) {
    int idx = blockIdx.x * blockDim.x + threadIdx.x;   // N*512 exact
    int n = idx >> 9, h = (idx & 511) >> 7;
    int nh = n * HEADS + h;
    float es = lrelu(als[nh] + ald[nh]);
    float alpha = __expf(es - maxf[nh]) / sum[nh];
    out1[idx] = alpha * h1[idx];
}
__global__ __launch_bounds__(128) void k_edge_agg1(const int* __restrict__ ei,
                                                   const float* __restrict__ h1,
                                                   const float* __restrict__ als,
                                                   const float* __restrict__ ald,
                                                   const float* __restrict__ maxf,
                                                   const float* __restrict__ sum,
                                                   float* __restrict__ out1) {
    int e = blockIdx.x;
    int t = threadIdx.x;                               // 0..127
    int s = ei[e], d = ei[N_EDGES + e];
#pragma unroll
    for (int h = 0; h < HEADS; ++h) {
        int nh = d * HEADS + h;
        float v = lrelu(als[s * HEADS + h] + ald[nh]);
        float alpha = __expf(v - maxf[nh]) / sum[nh];
        int ch = h * HID + t;
        atomicAdd(&out1[(size_t)d * HIDT + ch], alpha * h1[(size_t)s * HIDT + ch]);
    }
}
__global__ void k_finish1(const float* __restrict__ out1, const float* __restrict__ b1,
                          bf16_t* __restrict__ hactbf) {
    int idx = blockIdx.x * blockDim.x + threadIdx.x;   // N*512 exact
    int ch = idx & 511;
    float v = out1[idx] + b1[ch];
    v = v > 0.f ? v : (__expf(v) - 1.0f);              // ELU
    hactbf[idx] = (bf16_t)v;
}

// ---------------- layer-2 attention (H=1, C=30) ----------------
__global__ void k_attn2(const float* __restrict__ Z, const float* __restrict__ a_s,
                        const float* __restrict__ a_d,
                        float* __restrict__ als, float* __restrict__ ald) {
    int node = blockIdx.x * (blockDim.x >> 5) + (threadIdx.x >> 5);
    int lane = threadIdx.x & 31;
    if (node >= N_NODES) return;
    float vs = 0.f, vd = 0.f;
    if (lane < OUT_CH) {
        float v = Z[(size_t)node * OUT_CH + lane];
        vs = v * a_s[lane];
        vd = v * a_d[lane];
    }
    for (int off = 16; off; off >>= 1) {
        vs += __shfl_xor(vs, off, 32);
        vd += __shfl_xor(vd, off, 32);
    }
    if (lane == 0) { als[node] = vs; ald[node] = vd; }
}
__global__ void k_init_max2(const float* __restrict__ als, const float* __restrict__ ald,
                            unsigned* __restrict__ umax) {
    int n = blockIdx.x * blockDim.x + threadIdx.x;
    if (n >= N_NODES) return;
    umax[n] = fenc(lrelu(als[n] + ald[n]));
}
__global__ void k_edge_max2(const int* __restrict__ ei, const float* __restrict__ als,
                            const float* __restrict__ ald, unsigned* __restrict__ umax) {
    int e = blockIdx.x * blockDim.x + threadIdx.x;
    if (e >= N_EDGES) return;
    int s = ei[e], d = ei[N_EDGES + e];
    atomicMax(&umax[d], fenc(lrelu(als[s] + ald[d])));
}
__global__ void k_sum_init2(const unsigned* __restrict__ umax, const float* __restrict__ als,
                            const float* __restrict__ ald, float* __restrict__ maxf,
                            float* __restrict__ sum) {
    int n = blockIdx.x * blockDim.x + threadIdx.x;
    if (n >= N_NODES) return;
    float m = fdec(umax[n]);
    maxf[n] = m;
    sum[n] = __expf(lrelu(als[n] + ald[n]) - m);
}
__global__ void k_edge_sum2(const int* __restrict__ ei, const float* __restrict__ als,
                            const float* __restrict__ ald, const float* __restrict__ maxf,
                            float* __restrict__ sum) {
    int e = blockIdx.x * blockDim.x + threadIdx.x;
    if (e >= N_EDGES) return;
    int s = ei[e], d = ei[N_EDGES + e];
    atomicAdd(&sum[d], __expf(lrelu(als[s] + ald[d]) - maxf[d]));
}
__global__ void k_agg_init2(const float* __restrict__ Z, const float* __restrict__ als,
                            const float* __restrict__ ald, const float* __restrict__ maxf,
                            const float* __restrict__ sum, float* __restrict__ agg) {
    int idx = blockIdx.x * blockDim.x + threadIdx.x;
    if (idx >= N_NODES * OUT_CH) return;
    int n = idx / OUT_CH;
    float es = lrelu(als[n] + ald[n]);
    float alpha = __expf(es - maxf[n]) / sum[n];
    agg[idx] = alpha * Z[idx];
}
__global__ void k_edge_agg2(const int* __restrict__ ei, const float* __restrict__ Z,
                            const float* __restrict__ als, const float* __restrict__ ald,
                            const float* __restrict__ maxf, const float* __restrict__ sum,
                            float* __restrict__ agg) {
    int e = blockIdx.x * (blockDim.x >> 5) + (threadIdx.x >> 5);
    int lane = threadIdx.x & 31;
    if (e >= N_EDGES) return;
    int s = ei[e], d = ei[N_EDGES + e];
    float v = lrelu(als[s] + ald[d]);
    float alpha = __expf(v - maxf[d]) / sum[d];
    if (lane < OUT_CH)
        atomicAdd(&agg[(size_t)d * OUT_CH + lane], alpha * Z[(size_t)s * OUT_CH + lane]);
}
// z = agg + b2 -> d_out[z section]; also build zero-padded bf16 z for decoder GEMM
__global__ void k_finish2(const float* __restrict__ agg, const float* __restrict__ b2,
                          float* __restrict__ zout, bf16_t* __restrict__ zbf) {
    int idx = blockIdx.x * blockDim.x + threadIdx.x;   // N*32 exact
    int n = idx >> 5, c = idx & 31;
    float v = 0.f;
    if (c < OUT_CH) {
        v = agg[(size_t)n * OUT_CH + c] + b2[c];
        zout[(size_t)n * OUT_CH + c] = v;
    }
    zbf[idx] = (bf16_t)v;
}

// ---------------- host ----------------
static inline size_t align256(size_t x) { return (x + 255) & ~(size_t)255; }

extern "C" void kernel_launch(void* const* d_in, const int* in_sizes, int n_in,
                              void* d_out, int out_size, void* d_ws, size_t ws_size,
                              hipStream_t stream) {
    (void)in_sizes; (void)n_in; (void)out_size; (void)ws_size;
    const float* x      = (const float*)d_in[0];
    const int*   ei     = (const int*)  d_in[1];
    const float* W1     = (const float*)d_in[2];
    const float* a_src1 = (const float*)d_in[3];
    const float* a_dst1 = (const float*)d_in[4];
    const float* b1     = (const float*)d_in[5];
    const float* W2     = (const float*)d_in[6];
    const float* a_src2 = (const float*)d_in[7];
    const float* a_dst2 = (const float*)d_in[8];
    const float* b2     = (const float*)d_in[9];
    const float* Wd     = (const float*)d_in[10];
    const float* bd     = (const float*)d_in[11];

    float* xhat = (float*)d_out;                              // [N,512]
    float* zout = (float*)d_out + (size_t)N_NODES * IN_CH;    // [N,30]

    char* base = (char*)d_ws;
    size_t off = 0;
    auto alloc = [&](size_t bytes) -> void* {
        void* p = base + off; off += align256(bytes); return p;
    };
    bf16_t* xbf   = (bf16_t*)alloc((size_t)N_NODES * IN_CH * 2);   // reused as hactbf
    bf16_t* w1t   = (bf16_t*)alloc((size_t)IN_CH * HIDT * 2);
    bf16_t* w2t   = (bf16_t*)alloc((size_t)OUT_P * HIDT * 2);
    bf16_t* wdt   = (bf16_t*)alloc((size_t)IN_CH * OUT_P * 2);
    float*  h1    = (float*) alloc((size_t)N_NODES * HIDT * 4);
    float*  out1  = (float*) alloc((size_t)N_NODES * HIDT * 4);
    float*  als1  = (float*) alloc((size_t)N_NODES * HEADS * 4);
    float*  ald1  = (float*) alloc((size_t)N_NODES * HEADS * 4);
    unsigned* um1 = (unsigned*)alloc((size_t)N_NODES * HEADS * 4);
    float*  mx1   = (float*) alloc((size_t)N_NODES * HEADS * 4);
    float*  sm1   = (float*) alloc((size_t)N_NODES * HEADS * 4);
    float*  zpre  = (float*) alloc((size_t)N_NODES * OUT_CH * 4);
    float*  als2  = (float*) alloc((size_t)N_NODES * 4);
    float*  ald2  = (float*) alloc((size_t)N_NODES * 4);
    unsigned* um2 = (unsigned*)alloc((size_t)N_NODES * 4);
    float*  mx2   = (float*) alloc((size_t)N_NODES * 4);
    float*  sm2   = (float*) alloc((size_t)N_NODES * 4);
    float*  agg2  = (float*) alloc((size_t)N_NODES * OUT_CH * 4);
    bf16_t* zbf   = (bf16_t*)alloc((size_t)N_NODES * OUT_P * 2);
    bf16_t* hactbf = xbf;  // x no longer needed after gemm1

    const int MT = N_NODES / 16;  // 1250 tiles of rows

    // pack weights / inputs to bf16 WMMA layouts
    k_cvt_x   <<<(N_NODES * IN_CH) / 256, 256, 0, stream>>>(x, xbf);
    k_pack_w1t<<<(IN_CH * HIDT) / 256, 256, 0, stream>>>(W1, w1t);
    k_pack_w2t<<<(OUT_P * HIDT) / 256, 256, 0, stream>>>(W2, w2t);
    k_pack_wdt<<<(IN_CH * OUT_P) / 256, 256, 0, stream>>>(Wd, wdt);

    // layer 1: h1 = x @ W1
    k_gemm1<<<dim3(MT, HIDT / 64), 128, 0, stream>>>(xbf, w1t, h1);

    // layer-1 attention softmax + aggregation (self-loops analytic)
    k_attn1    <<<N_NODES / 4, 128, 0, stream>>>(h1, a_src1, a_dst1, als1, ald1);
    k_init_max1<<<(N_NODES * HEADS + 255) / 256, 256, 0, stream>>>(als1, ald1, um1);
    k_edge_max1<<<(N_EDGES + 255) / 256, 256, 0, stream>>>(ei, als1, ald1, um1);
    k_sum_init1<<<(N_NODES * HEADS + 255) / 256, 256, 0, stream>>>(um1, als1, ald1, mx1, sm1);
    k_edge_sum1<<<(N_EDGES + 255) / 256, 256, 0, stream>>>(ei, als1, ald1, mx1, sm1);
    k_agg_init1<<<(N_NODES * HIDT) / 256, 256, 0, stream>>>(h1, als1, ald1, mx1, sm1, out1);
    k_edge_agg1<<<N_EDGES, 128, 0, stream>>>(ei, h1, als1, ald1, mx1, sm1, out1);
    k_finish1  <<<(N_NODES * HIDT) / 256, 256, 0, stream>>>(out1, b1, hactbf);

    // layer 2: zpre = elu(h) @ W2
    k_gemm2<<<MT, 64, 0, stream>>>(hactbf, w2t, zpre);

    k_attn2    <<<N_NODES / 8, 256, 0, stream>>>(zpre, a_src2, a_dst2, als2, ald2);
    k_init_max2<<<(N_NODES + 255) / 256, 256, 0, stream>>>(als2, ald2, um2);
    k_edge_max2<<<(N_EDGES + 255) / 256, 256, 0, stream>>>(ei, als2, ald2, um2);
    k_sum_init2<<<(N_NODES + 255) / 256, 256, 0, stream>>>(um2, als2, ald2, mx2, sm2);
    k_edge_sum2<<<(N_EDGES + 255) / 256, 256, 0, stream>>>(ei, als2, ald2, mx2, sm2);
    k_agg_init2<<<(N_NODES * OUT_CH + 255) / 256, 256, 0, stream>>>(zpre, als2, ald2, mx2, sm2, agg2);
    k_edge_agg2<<<N_EDGES / 8, 256, 0, stream>>>(ei, zpre, als2, ald2, mx2, sm2, agg2);
    k_finish2  <<<(N_NODES * OUT_P) / 256, 256, 0, stream>>>(agg2, b2, zout, zbf);

    // decoder: x_hat = z @ Wd + bd
    k_gemm3<<<dim3(MT, IN_CH / 64), 128, 0, stream>>>(zbf, wdt, bd, xhat);
}
